// SparseMoE_75488345195332
// MI455X (gfx1250) — matrix operand order repaired and checked by reading, not compiled, not run on hardware
//
#include <hip/hip_runtime.h>
#include <hip/hip_bf16.h>
#include <math.h>

typedef __bf16 bf16;
typedef __attribute__((ext_vector_type(16))) __bf16 v16bf;
typedef __attribute__((ext_vector_type(8)))  __bf16 v8bf;
typedef __attribute__((ext_vector_type(8)))  float  v8f;

#define DEVFN static __device__ __forceinline__

constexpr int   N_  = 16384, D_ = 256, H_ = 512, E_ = 16, K_ = 2;
constexpr float LBW = 0.01f, EPS_ = 1e-5f;
constexpr int   NK  = N_ * K_;                 // 32768 dispatch entries
constexpr int   MAXTILES = NK / 16 + E_;       // 2064 worst-case 16-row tiles

// ---- output layout (float elements, concatenated tuple) ----
constexpr size_t OUT_OFF  = 0;                         // [N,D] f32
constexpr size_t LB_OFF   = (size_t)N_ * D_;           // scalar
constexpr size_t PROB_OFF = LB_OFF + 1;                // [N,E] f32
constexpr size_t TOPI_OFF = PROB_OFF + (size_t)N_*E_;  // [N,K] int32 bits

// ---- workspace layout (bytes) ----
constexpr size_t W1T_OFF   = 0;                                   // bf16 [E][H][D]
constexpr size_t W2T_OFF   = W1T_OFF + (size_t)E_*D_*H_*2;        // bf16 [E][D][H]
constexpr size_t CNT_OFF   = W2T_OFF + (size_t)E_*H_*D_*2;        // int  [E]
constexpr size_t OFFS_OFF  = CNT_OFF  + 128;                      // int  [E+1]
constexpr size_t CUR_OFF   = OFFS_OFF + 128;                      // int  [E]
constexpr size_t PSUM_OFF  = CUR_OFF  + 128;                      // f32  [E]
constexpr size_t NT_OFF    = PSUM_OFF + 128;                      // int  numTiles
constexpr size_t TILEE_OFF = NT_OFF   + 128;                      // int  [MAXTILES]
constexpr size_t TILEB_OFF = TILEE_OFF + ((MAXTILES*4+127)/128)*128;
constexpr size_t ETOK_OFF  = TILEB_OFF + ((MAXTILES*4+127)/128)*128; // int [NK]
constexpr size_t EW_OFF    = ETOK_OFF + (size_t)NK*4;             // f32 [NK]
constexpr size_t TOPW_OFF  = EW_OFF   + (size_t)NK*4;             // f32 [N*K]
// total ~8.8 MB of ws

DEVFN v16bf pack16(v8bf lo, v8bf hi) {
  return __builtin_shufflevector(lo, hi, 0,1,2,3,4,5,6,7,8,9,10,11,12,13,14,15);
}
DEVFN v8f wmma_bf16(v16bf a, v16bf b, v8f c) {
  return __builtin_amdgcn_wmma_f32_16x16x32_bf16(false, a, false, b, (short)0, c,
                                                 false, false);
}
DEVFN float gelu_erf(float v) {
  return 0.5f * v * (1.0f + erff(v * 0.70710678118654752f));
}

// ---------------- kernel 1: zero per-launch metadata ----------------
__global__ void moe_zero_meta(int* counts, float* psum) {
  if (threadIdx.x < E_) { counts[threadIdx.x] = 0; psum[threadIdx.x] = 0.0f; }
}

// ---------------- kernel 2: convert + transpose weights to bf16 -----
// w1t[e][h][d] = W1[e][d][h]  (B operand of GEMM1: K=d contiguous per column h)
// w2t[e][d][h] = W2[e][h][d]  (B operand of GEMM2: K=h contiguous per column d)
__global__ void moe_convert_w(const float* __restrict__ W1,
                              const float* __restrict__ W2,
                              bf16* __restrict__ w1t, bf16* __restrict__ w2t) {
  int idx = blockIdx.x * blockDim.x + threadIdx.x;
  const int tot = E_ * D_ * H_;
  if (idx < tot) {
    int e = idx / (H_ * D_), r = idx % (H_ * D_);
    int h = r / D_, d = r % D_;
    w1t[idx] = (bf16)W1[((size_t)e * D_ + d) * H_ + h];
  } else if (idx < 2 * tot) {
    int j = idx - tot;
    int e = j / (D_ * H_), r = j % (D_ * H_);
    int d = r / H_, h = r % H_;
    w2t[j] = (bf16)W2[((size_t)e * H_ + h) * D_ + d];
  }
}

// ---------------- kernel 3: accumulator = x (residual) --------------
__global__ void moe_init_acc(const float4* __restrict__ x, float4* __restrict__ acc,
                             int n4) {
  int i = blockIdx.x * blockDim.x + threadIdx.x;
  if (i < n4) acc[i] = x[i];
}

// ---------------- kernel 4: router (wave32 per token) ---------------
__global__ void __launch_bounds__(256) moe_router(
    const float* __restrict__ x, const float* __restrict__ Wr,
    float* __restrict__ probs_out, int* __restrict__ topi_out,
    float* __restrict__ topw_out, int* __restrict__ counts,
    float* __restrict__ psum) {
  __shared__ float wr[E_ * D_];
  __shared__ float ps[E_];
  for (int i = threadIdx.x; i < E_ * D_; i += blockDim.x) wr[i] = Wr[i];
  if (threadIdx.x < E_) ps[threadIdx.x] = 0.0f;
  __syncthreads();

  const int wave = threadIdx.x >> 5, lane = threadIdx.x & 31;
  const int tok = blockIdx.x * 8 + wave;

  float acc[E_];
#pragma unroll
  for (int e = 0; e < E_; ++e) acc[e] = 0.0f;
  const float* xr = x + (size_t)tok * D_;
#pragma unroll
  for (int i = 0; i < D_ / 32; ++i) {
    int d = lane + 32 * i;
    float xv = xr[d];
#pragma unroll
    for (int e = 0; e < E_; ++e) acc[e] += xv * wr[e * D_ + d];
  }
#pragma unroll
  for (int m = 16; m >= 1; m >>= 1)
#pragma unroll
    for (int e = 0; e < E_; ++e) acc[e] += __shfl_xor(acc[e], m, 32);

  // softmax over 16 logits (all lanes redundantly)
  float mx = acc[0];
#pragma unroll
  for (int e = 1; e < E_; ++e) mx = fmaxf(mx, acc[e]);
  float p[E_], s = 0.0f;
#pragma unroll
  for (int e = 0; e < E_; ++e) { p[e] = expf(acc[e] - mx); s += p[e]; }
  float inv = 1.0f / s;
#pragma unroll
  for (int e = 0; e < E_; ++e) p[e] *= inv;

  // top-2 (lowest index wins ties, like lax.top_k)
  int i0 = 0;
#pragma unroll
  for (int e = 1; e < E_; ++e) if (p[e] > p[i0]) i0 = e;
  int i1 = (i0 == 0) ? 1 : 0;
#pragma unroll
  for (int e = 0; e < E_; ++e) if (e != i0 && p[e] > p[i1]) i1 = e;
  float rn = 1.0f / (p[i0] + p[i1] + 1e-9f);
  float w0 = p[i0] * rn, w1 = p[i1] * rn;

  if (lane == 0 && tok < N_) {
    float* pr = probs_out + (size_t)tok * E_;
#pragma unroll
    for (int e = 0; e < E_; ++e) { pr[e] = p[e]; atomicAdd(&ps[e], p[e]); }
    topi_out[tok * K_ + 0] = i0;
    topi_out[tok * K_ + 1] = i1;
    topw_out[tok * K_ + 0] = w0;
    topw_out[tok * K_ + 1] = w1;
    atomicAdd(&counts[i0], 1);
    atomicAdd(&counts[i1], 1);
  }
  __syncthreads();
  if (threadIdx.x < E_) atomicAdd(&psum[threadIdx.x], ps[threadIdx.x]);
}

// ---------------- kernel 5: scan offsets, build tiles, lb_loss ------
__global__ void moe_scan(const int* __restrict__ counts, int* __restrict__ offsets,
                         int* __restrict__ cursors, const float* __restrict__ psum,
                         int* __restrict__ tileE, int* __restrict__ tileB,
                         int* __restrict__ numTiles, float* __restrict__ lb_out) {
  if (threadIdx.x == 0) {
    int off = 0, nt = 0;
    for (int e = 0; e < E_; ++e) {
      offsets[e] = off; cursors[e] = off;
      int c = counts[e];
      int t = (c + 15) >> 4;
      for (int i = 0; i < t; ++i) { tileE[nt] = e; tileB[nt] = off + i * 16; ++nt; }
      off += c;
    }
    offsets[E_] = off;
    *numTiles = nt;
    float lb = 0.0f;
    for (int e = 0; e < E_; ++e)
      lb += ((float)counts[e] / ((float)NK + 1e-9f)) * (psum[e] / (float)N_);
    *lb_out = LBW * (float)E_ * lb;
  }
}

// ---------------- kernel 6: scatter tokens into expert lists --------
__global__ void moe_scatter(const int* __restrict__ topi, const float* __restrict__ topw,
                            int* __restrict__ cursors, int* __restrict__ etok,
                            float* __restrict__ ew) {
  int tok = blockIdx.x * blockDim.x + threadIdx.x;
  if (tok >= N_) return;
#pragma unroll
  for (int k = 0; k < K_; ++k) {
    int e = topi[tok * K_ + k];
    int pos = atomicAdd(&cursors[e], 1);
    etok[pos] = tok;
    ew[pos]   = topw[tok * K_ + k];
  }
}

// ---------------- kernel 7: expert FFN (WMMA bf16) ------------------
// block = one (expert, 16-token tile); 128 threads = 4 waves.
__global__ void __launch_bounds__(128) moe_ffn(
    const bf16* __restrict__ w1t, const bf16* __restrict__ w2t,
    const int* __restrict__ etok, const float* __restrict__ ew,
    const int* __restrict__ tileE, const int* __restrict__ tileB,
    const int* __restrict__ numTiles, const int* __restrict__ offsets,
    const float* __restrict__ x, float* __restrict__ acc) {
  if ((int)blockIdx.x >= *numTiles) return;

  __shared__ __attribute__((aligned(32))) bf16 xa[16 * D_];  // A of GEMM1 (8KB)
  __shared__ __attribute__((aligned(32))) bf16 hl[16 * H_];  // A of GEMM2 (16KB)
  __shared__ int   toks[16];
  __shared__ float gw[16];

  const int e = tileE[blockIdx.x], base = tileB[blockIdx.x];
  int cnt = offsets[e + 1] - base; if (cnt > 16) cnt = 16;
  const int tid = threadIdx.x;

  if (tid < 16) {
    if (tid < cnt) { toks[tid] = etok[base + tid]; gw[tid] = ew[base + tid]; }
    else           { toks[tid] = etok[base];       gw[tid] = 0.0f; }  // padded row: weight 0
  }
  __syncthreads();
  for (int i = tid; i < 16 * D_; i += 128) {            // gather + cvt x tile -> bf16
    int row = i >> 8, col = i & (D_ - 1);
    xa[i] = (bf16)x[(size_t)toks[row] * D_ + col];
  }
  __syncthreads();

  const int wave = tid >> 5, lane = tid & 31;
  const int rc = lane & 15;         // A-row / B-col / C-col index
  const int half = lane >> 4;       // upper/lower K half

  // ---- GEMM1: h[16xH] = xa[16xD] @ W1_e[DxH], gelu, -> hl (bf16) ----
  const bf16* w1e = w1t + (size_t)e * H_ * D_;
#pragma unroll
  for (int t = 0; t < 8; ++t) {                      // 8 N-tiles per wave (H=512)
    const int hbase = (wave * 8 + t) * 16;
    const int hcol  = hbase + rc;
    const bf16* bp  = w1e + (size_t)hcol * D_;
    v8f c = {};
#pragma unroll
    for (int k0 = 0; k0 < D_; k0 += 32) {
      v8bf alo = *(const v8bf*)(&xa[rc * D_ + k0 + half * 8]);
      v8bf ahi = *(const v8bf*)(&xa[rc * D_ + k0 + 16 + half * 8]);
      v8bf blo = *(const v8bf*)(bp + k0 + half * 8);
      v8bf bhi = *(const v8bf*)(bp + k0 + 16 + half * 8);
      c = wmma_bf16(pack16(alo, ahi), pack16(blo, bhi), c);
    }
#pragma unroll
    for (int r = 0; r < 8; ++r) {                    // exact-erf GELU, store A2
      int m = r + half * 8;
      hl[m * H_ + hcol] = (bf16)gelu_erf(c[r]);
    }
  }
  __syncthreads();

  // ---- GEMM2: y[16xD] = hl[16xH] @ W2_e[HxD], gated atomic combine ----
  const bf16* w2e = w2t + (size_t)e * D_ * H_;
#pragma unroll
  for (int t = 0; t < 4; ++t) {                      // 4 N-tiles per wave (D=256)
    const int dbase = (wave * 4 + t) * 16;
    const int dcol  = dbase + rc;
    const bf16* bp  = w2e + (size_t)dcol * H_;
    v8f c = {};
#pragma unroll
    for (int k0 = 0; k0 < H_; k0 += 32) {
      v8bf alo = *(const v8bf*)(&hl[rc * H_ + k0 + half * 8]);
      v8bf ahi = *(const v8bf*)(&hl[rc * H_ + k0 + 16 + half * 8]);
      v8bf blo = *(const v8bf*)(bp + k0 + half * 8);
      v8bf bhi = *(const v8bf*)(bp + k0 + 16 + half * 8);
      c = wmma_bf16(pack16(alo, ahi), pack16(blo, bhi), c);
    }
#pragma unroll
    for (int r = 0; r < 8; ++r) {
      int m = r + half * 8;
      atomicAdd(&acc[(size_t)toks[m] * D_ + dcol], gw[m] * c[r]);
    }
  }
}

// ---------------- kernel 8: in-place LayerNorm ----------------------
__global__ void __launch_bounds__(256) moe_ln(float* __restrict__ out,
                                              const float* __restrict__ gamma,
                                              const float* __restrict__ beta) {
  __shared__ float red[256];
  const int tok = blockIdx.x, d = threadIdx.x;
  float z = out[(size_t)tok * D_ + d];
  red[d] = z; __syncthreads();
  for (int s = 128; s; s >>= 1) { if (d < s) red[d] += red[d + s]; __syncthreads(); }
  float mu = red[0] * (1.0f / D_);
  __syncthreads();
  float dv = z - mu;
  red[d] = dv * dv; __syncthreads();
  for (int s = 128; s; s >>= 1) { if (d < s) red[d] += red[d + s]; __syncthreads(); }
  float var = red[0] * (1.0f / D_);
  out[(size_t)tok * D_ + d] = dv * rsqrtf(var + EPS_) * gamma[d] + beta[d];
}

extern "C" void kernel_launch(void* const* d_in, const int* in_sizes, int n_in,
                              void* d_out, int out_size, void* d_ws, size_t ws_size,
                              hipStream_t stream) {
  (void)in_sizes; (void)n_in; (void)out_size; (void)ws_size;
  const float* x     = (const float*)d_in[0];
  const float* Wr    = (const float*)d_in[1];
  const float* W1    = (const float*)d_in[2];
  const float* W2    = (const float*)d_in[3];
  const float* gamma = (const float*)d_in[4];
  const float* beta  = (const float*)d_in[5];
  float* out = (float*)d_out;
  char*  ws  = (char*)d_ws;

  bf16*  w1t     = (bf16*) (ws + W1T_OFF);
  bf16*  w2t     = (bf16*) (ws + W2T_OFF);
  int*   counts  = (int*)  (ws + CNT_OFF);
  int*   offsets = (int*)  (ws + OFFS_OFF);
  int*   cursors = (int*)  (ws + CUR_OFF);
  float* psum    = (float*)(ws + PSUM_OFF);
  int*   numT    = (int*)  (ws + NT_OFF);
  int*   tileE   = (int*)  (ws + TILEE_OFF);
  int*   tileB   = (int*)  (ws + TILEB_OFF);
  int*   etok    = (int*)  (ws + ETOK_OFF);
  float* ewt     = (float*)(ws + EW_OFF);
  float* topw    = (float*)(ws + TOPW_OFF);
  int*   topi    = (int*)  (out + TOPI_OFF);

  moe_zero_meta<<<1, 64, 0, stream>>>(counts, psum);
  moe_convert_w<<<(2 * E_ * D_ * H_ + 255) / 256, 256, 0, stream>>>(W1, W2, w1t, w2t);
  moe_init_acc<<<(N_ * D_ / 4 + 255) / 256, 256, 0, stream>>>(
      (const float4*)x, (float4*)out, N_ * D_ / 4);
  moe_router<<<N_ / 8, 256, 0, stream>>>(x, Wr, out + PROB_OFF, topi, topw, counts, psum);
  moe_scan<<<1, 32, 0, stream>>>(counts, offsets, cursors, psum, tileE, tileB, numT,
                                 out + LB_OFF);
  moe_scatter<<<(N_ + 255) / 256, 256, 0, stream>>>(topi, topw, cursors, etok, ewt);
  moe_ffn<<<MAXTILES, 128, 0, stream>>>(w1t, w2t, etok, ewt, tileE, tileB, numT,
                                        offsets, x, out);
  moe_ln<<<N_, 256, 0, stream>>>(out, gamma, beta);
}
